// mulGCN_57526791962625
// MI455X (gfx1250) — matrix operand order repaired and checked by reading, not compiled, not run on hardware
//
#include <hip/hip_runtime.h>

#define NN 100000   // nodes
#define NE 1200000  // edges
#define NG 256      // graphs
// D = 64, G_FEAT = 128

typedef _Float16 v16h __attribute__((ext_vector_type(16)));
typedef float    v8f  __attribute__((ext_vector_type(8)));
typedef unsigned short f16raw;

#define NTILES_TOTAL 40        // Wg(8) Wr(8) Wi(8) Wo(16)
#define TILE_G 0
#define TILE_R 8
#define TILE_I 16
#define TILE_O 24
#define LSTRIDE 20             // f16 per lane in LDS weight cache (16 used + 4 pad)

#define WMMA_F16(a, b, c) \
    __builtin_amdgcn_wmma_f32_16x16x32_f16(false, (a), false, (b), (short)0, (c), false, false)

// ---------------------------------------------------------------------------
// A-operand builders: 16x32 f16 A layout (wave32).
// lane: M = lane&15, half = lane>>4.
// v16h elements 0..7  = A[M][Kbase + half*8 + i]
// v16h elements 8..15 = A[M][Kbase + 16 + half*8 + i]
// ---------------------------------------------------------------------------
__device__ __forceinline__ v16h a_from_f32(const float* __restrict__ p) {
    v16h a;
#pragma unroll
    for (int i = 0; i < 8; ++i) {
        a[i]     = (_Float16)p[i];
        a[i + 8] = (_Float16)p[i + 16];
    }
    return a;
}

__device__ __forceinline__ v16h a_from_lds(const _Float16* p) {
    v16h a;
#pragma unroll
    for (int i = 0; i < 8; ++i) {
        a[i]     = p[i];
        a[i + 8] = p[i + 16];
    }
    return a;
}

// B tile fetch from the per-block LDS weight cache (conflict-free: 10*L mod 64
// is injective over the 32 lanes with the 40-byte lane stride).
__device__ __forceinline__ v16h b_tile_lds(const _Float16* Wl, int tile, int lane) {
    const _Float16* p = Wl + ((size_t)tile * 32 + lane) * LSTRIDE;
    v16h b;
#pragma unroll
    for (int i = 0; i < 16; ++i) b[i] = p[i];
    return b;
}

// ---------------------------------------------------------------------------
// Kernel 0: zero agg + g_feats workspace (float4 granularity)
// ---------------------------------------------------------------------------
__global__ void zero4_kernel(float4* __restrict__ p, int n4) {
    int t = blockIdx.x * blockDim.x + threadIdx.x;
    if (t < n4) p[t] = make_float4(0.f, 0.f, 0.f, 0.f);
}

// ---------------------------------------------------------------------------
// Kernel 1: pre-pack f32 weights -> f16 WMMA B layout (contiguous 40-tile blob).
// B layout (32x16 f16, wave32): lane holds column N = lane&15,
// K = (e>>3)*16 + (lane>>4)*8 + (e&7)  (e = element 0..15), for K-step ks: K += ks*32.
// ---------------------------------------------------------------------------
__global__ void prepack_kernel(const float* __restrict__ Wg, const float* __restrict__ Wr,
                               const float* __restrict__ Wi, const float* __restrict__ Wo,
                               f16raw* __restrict__ Pall) {
    int t = blockIdx.x * blockDim.x + threadIdx.x;
    if (t >= NTILES_TOTAL * 32) return;
    int tile = t >> 5, lane = t & 31;
    const float* W; int Ndim; int lt;
    if      (tile < TILE_R) { W = Wg; Ndim = 64;  lt = tile - TILE_G; }
    else if (tile < TILE_I) { W = Wr; Ndim = 64;  lt = tile - TILE_R; }
    else if (tile < TILE_O) { W = Wi; Ndim = 64;  lt = tile - TILE_I; }
    else                    { W = Wo; Ndim = 128; lt = tile - TILE_O; }
    int ntiles = Ndim / 16;
    int ks = lt / ntiles, nt = lt % ntiles;
    int half = lane >> 4, n = lane & 15;
    f16raw* out = Pall + ((size_t)tile * 32 + lane) * 16;
#pragma unroll
    for (int e = 0; e < 16; ++e) {
        _Float16 hv = (_Float16)W[(size_t)(ks * 32 + (e >> 3) * 16 + half * 8 + (e & 7)) * Ndim + nt * 16 + n];
        out[e] = *(const f16raw*)&hv;
    }
}

// ---------------------------------------------------------------------------
// Kernel 2: edge scatter  agg[dst] += node_feats[src]   (4 threads / edge)
// ---------------------------------------------------------------------------
__global__ void scatter_kernel(const float* __restrict__ nf, const int* __restrict__ src,
                               const int* __restrict__ dst, float* __restrict__ agg) {
    long long t = (long long)blockIdx.x * blockDim.x + threadIdx.x;
    int e = (int)(t >> 2);
    if (e >= NE) return;
    int part = (int)(t & 3);
    int s = src[e], d = dst[e];
    const float4* sp = (const float4*)(nf + (size_t)s * 64 + part * 16);
    float* dp = agg + (size_t)d * 64 + part * 16;
#pragma unroll
    for (int c = 0; c < 4; ++c) {
        float4 v = sp[c];
        atomicAdd(dp + 4 * c + 0, v.x);
        atomicAdd(dp + 4 * c + 1, v.y);
        atomicAdd(dp + 4 * c + 2, v.z);
        atomicAdd(dp + 4 * c + 3, v.w);
    }
}

// ---------------------------------------------------------------------------
// Kernel 3: fused per-node pipeline (4 waves/block, 16 nodes/wave).
// All packed weights staged into LDS once per block; WMMA B feeds from LDS.
//   h1 = relu(agg@Wg + bg) + relu(x@Wr + br)
//   h2 = relu(h1@Wi + bi)
//   y  = h2@Wo + bo ;  g_feats[gid[node]] += y   (atomic)
// ---------------------------------------------------------------------------
__global__ __launch_bounds__(128) void fused_node_kernel(
    const float* __restrict__ agg, const float* __restrict__ nf,
    const int*   __restrict__ gid,
    const f16raw* __restrict__ Pall,
    const float* __restrict__ bg, const float* __restrict__ br,
    const float* __restrict__ bi, const float* __restrict__ bo,
    float* __restrict__ gf) {
    __shared__ __align__(16) _Float16 Wlds[NTILES_TOTAL * 32 * LSTRIDE]; // 51.2 KB
    __shared__ __align__(16) _Float16 stage[4][16 * 72];                 //  9.2 KB
    const int wave = threadIdx.x >> 5;
    const int lane = threadIdx.x & 31;
    const int half = lane >> 4;
    const int m    = lane & 15;
    const int row0 = (blockIdx.x * 4 + wave) * 16;
    _Float16* L = &stage[wave][0];

    // ---- Cooperative copy: packed weights (global, tile-major) -> LDS ----
    for (int tl = threadIdx.x; tl < NTILES_TOTAL * 32; tl += 128) {
        const _Float16* s = (const _Float16*)Pall + (size_t)tl * 16;
        _Float16* d = Wlds + (size_t)tl * LSTRIDE;
#pragma unroll
        for (int e = 0; e < 16; ++e) d[e] = s[e];
    }

    // ---- Load A operands (agg rows & node_feats rows), f32 -> f16 A layout ----
    int r = row0 + m;
    if (r >= NN) r = NN - 1;                 // clamp; pad rows masked at readout
    const float* pa = agg + (size_t)r * 64;
    const float* px = nf  + (size_t)r * 64;
    v16h aA0 = a_from_f32(pa + half * 8);
    v16h aA1 = a_from_f32(pa + 32 + half * 8);
    v16h aX0 = a_from_f32(px + half * 8);
    v16h aX1 = a_from_f32(px + 32 + half * 8);

    __syncthreads();   // weights in LDS visible to all waves

    // ---- GEMM1 (+ residual branch): h1 -> LDS staging ----
#pragma unroll
    for (int nt = 0; nt < 4; ++nt) {
        v8f c = {};
        c = WMMA_F16(aA0, b_tile_lds(Wlds, TILE_G + 0 * 4 + nt, lane), c);
        c = WMMA_F16(aA1, b_tile_lds(Wlds, TILE_G + 1 * 4 + nt, lane), c);
        v8f d = {};
        d = WMMA_F16(aX0, b_tile_lds(Wlds, TILE_R + 0 * 4 + nt, lane), d);
        d = WMMA_F16(aX1, b_tile_lds(Wlds, TILE_R + 1 * 4 + nt, lane), d);
        float cb = bg[nt * 16 + m], db = br[nt * 16 + m];
#pragma unroll
        for (int i = 0; i < 8; ++i) {
            float h = fmaxf(c[i] + cb, 0.f) + fmaxf(d[i] + db, 0.f);
            L[(i + 8 * half) * 72 + nt * 16 + m] = (_Float16)h;
        }
    }
    __syncthreads();

    // ---- GEMM2: h2 = relu(h1@Wi + bi) -> LDS staging ----
    v16h h10 = a_from_lds(L + m * 72 + half * 8);
    v16h h11 = a_from_lds(L + m * 72 + 32 + half * 8);
    v8f t2[4];
#pragma unroll
    for (int nt = 0; nt < 4; ++nt) {
        v8f c = {};
        c = WMMA_F16(h10, b_tile_lds(Wlds, TILE_I + 0 * 4 + nt, lane), c);
        c = WMMA_F16(h11, b_tile_lds(Wlds, TILE_I + 1 * 4 + nt, lane), c);
        t2[nt] = c;
    }
    __syncthreads();   // everyone done reading h1 before overwrite
#pragma unroll
    for (int nt = 0; nt < 4; ++nt) {
        float cb = bi[nt * 16 + m];
#pragma unroll
        for (int i = 0; i < 8; ++i)
            L[(i + 8 * half) * 72 + nt * 16 + m] = (_Float16)fmaxf(t2[nt][i] + cb, 0.f);
    }
    __syncthreads();

    // ---- GEMM3: y = h2@Wo + bo, scatter into graph features ----
    v16h h20 = a_from_lds(L + m * 72 + half * 8);
    v16h h21 = a_from_lds(L + m * 72 + 32 + half * 8);
    v8f y[8];
#pragma unroll
    for (int nt = 0; nt < 8; ++nt) {
        v8f c = {};
        c = WMMA_F16(h20, b_tile_lds(Wlds, TILE_O + 0 * 8 + nt, lane), c);
        c = WMMA_F16(h21, b_tile_lds(Wlds, TILE_O + 1 * 8 + nt, lane), c);
        y[nt] = c;
    }
    float bov[8];
#pragma unroll
    for (int nt = 0; nt < 8; ++nt) bov[nt] = bo[nt * 16 + m];
#pragma unroll
    for (int i = 0; i < 8; ++i) {
        int rr = row0 + i + 8 * half;
        if (rr < NN) {
            int g = gid[rr];
            float* dp = gf + (size_t)g * 128 + m;
#pragma unroll
            for (int nt = 0; nt < 8; ++nt)
                atomicAdd(dp + nt * 16, y[nt][i] + bov[nt]);
        }
    }
}

// ---------------------------------------------------------------------------
// Kernel 4: out[g] = g_feats[g] . W_pred + b_pred   (256 x 128 x 1)
// ---------------------------------------------------------------------------
__global__ void predict_kernel(const float* __restrict__ gf, const float* __restrict__ Wp,
                               const float* __restrict__ bp, float* __restrict__ out) {
    int g = blockIdx.x * blockDim.x + threadIdx.x;
    if (g >= NG) return;
    const float* row = gf + (size_t)g * 128;
    float s = bp[0];
#pragma unroll 8
    for (int k = 0; k < 128; ++k) s += row[k] * Wp[k];
    out[g] = s;
}

// ---------------------------------------------------------------------------
extern "C" void kernel_launch(void* const* d_in, const int* in_sizes, int n_in,
                              void* d_out, int out_size, void* d_ws, size_t ws_size,
                              hipStream_t stream) {
    (void)in_sizes; (void)n_in; (void)out_size; (void)ws_size;
    const float* nf  = (const float*)d_in[0];
    // d_in[1] edge_feats: unused by the reference
    const int*   src = (const int*)d_in[2];
    const int*   dst = (const int*)d_in[3];
    const int*   gid = (const int*)d_in[4];
    const float* Wg  = (const float*)d_in[5];
    const float* bg  = (const float*)d_in[6];
    const float* Wr  = (const float*)d_in[7];
    const float* br  = (const float*)d_in[8];
    const float* Wi  = (const float*)d_in[9];
    const float* bi  = (const float*)d_in[10];
    const float* Wo  = (const float*)d_in[11];
    const float* bo  = (const float*)d_in[12];
    const float* Wp  = (const float*)d_in[13];
    const float* bp  = (const float*)d_in[14];
    float* out = (float*)d_out;

    char* ws = (char*)d_ws;
    float*  agg  = (float*)ws;                                      // 100000*64 f32
    float*  gf   = (float*)(ws + (size_t)NN * 64 * 4);              // 256*128 f32
    f16raw* Pall = (f16raw*)(ws + (size_t)NN * 64 * 4 + (size_t)NG * 128 * 4); // 40 KB

    int n4 = (NN * 64 + NG * 128) / 4;
    zero4_kernel<<<(n4 + 255) / 256, 256, 0, stream>>>((float4*)ws, n4);
    prepack_kernel<<<(NTILES_TOTAL * 32 + 255) / 256, 256, 0, stream>>>(Wg, Wr, Wi, Wo, Pall);
    scatter_kernel<<<(NE * 4 + 255) / 256, 256, 0, stream>>>(nf, src, dst, agg);
    fused_node_kernel<<<(NN + 63) / 64, 128, 0, stream>>>(agg, nf, gid, Pall,
                                                          bg, br, bi, bo, gf);
    predict_kernel<<<1, 256, 0, stream>>>(gf, Wp, bp, out);
}
